// MPDNN_25786983645311
// MI455X (gfx1250) — compile-verified, matrix operand order checked by reading, not run on hardware
//
#include <hip/hip_runtime.h>
#include <hip/hip_bf16.h>
#include <math.h>

// ---------------------------------------------------------------------------
// Problem constants (match reference)
// ---------------------------------------------------------------------------
#define BB 16
#define NN 128
#define FF 128
#define KK 32
#define N_CONV 3
#define RC 6.0f
#define LOG2F_ 0.69314718055994530942f
#define INV2W2 8.0f            // 1/(2*WIDTH^2), WIDTH=0.25
#define MU_STEP (5.5f / 31.0f)

typedef __attribute__((ext_vector_type(16))) _Float16 v16h;
typedef __attribute__((ext_vector_type(2)))  _Float16 v2h;
typedef __attribute__((ext_vector_type(8)))  float    v8f;

// ---------------------------------------------------------------------------
// Helpers
// ---------------------------------------------------------------------------
// shifted softplus with HW transcendentals (v_exp_f32 / v_log_f32).
// Stable: max(x,0) + log(1+exp(-|x|)) - log2; arg of log is in (1,2].
__device__ __forceinline__ float sspf(float x) {
  float ax = fabsf(x);
  return fmaxf(x, 0.0f) + __logf(1.0f + __expf(-ax)) - LOG2F_;
}

// K index held by operand element t for a lane in the low (hi=0) or high
// (hi=1) half-wave, per the CDNA5 16-bit 16x32 A/B fragment layout:
//  lanes 0-15 : v0..v3 -> K 0..7,  v4..v7 -> K 16..23
//  lanes16-31 : v0..v3 -> K 8..15, v4..v7 -> K 24..31
__device__ __forceinline__ int kmap(int t, int hi) {
  int k = (t < 8) ? t : (t + 8);
  return k + (hi ? 8 : 0);
}

__device__ __forceinline__ v8f wmma_f16(v16h a, v16h b, v8f c) {
  return __builtin_amdgcn_wmma_f32_16x16x32_f16(
      /*neg_a=*/false, a, /*neg_b=*/false, b,
      /*c_mod=*/(short)0, c, /*reuse_a=*/false, /*reuse_b=*/false);
}

// Async global->LDS copy of 16 bytes per lane (ASYNCcnt-tracked).
// lds_dst: generic pointer to __shared__; low 32 bits are the wave-relative
// LDS byte address (ISA 10.2: flat LDS addr = addr[31:0]).
__device__ __forceinline__ void async_copy_b128(void* lds_dst, const void* gsrc) {
  unsigned int lds = (unsigned int)(unsigned long long)lds_dst;
  unsigned long long ga = (unsigned long long)gsrc;
  asm volatile("global_load_async_to_lds_b128 %0, %1, off"
               :: "v"(lds), "v"(ga) : "memory");
}
__device__ __forceinline__ void wait_async0() {
  asm volatile("s_wait_asynccnt 0" ::: "memory");
}

// ---------------------------------------------------------------------------
// Weight repack: f32 [Krows x Ncols] row-major -> f16 B-operand images
// Bop[(tile*Kchunks+chunk)*512 + lane*16 + t] = W[kmap(t,hi)+32*chunk][tile*16+lo]
// ---------------------------------------------------------------------------
__global__ void k_prep_bop(const float* __restrict__ W, _Float16* __restrict__ Bop,
                           int Krows, int Ncols, int Kchunks) {
  int id = blockIdx.x;                    // tile*Kchunks + chunk
  int chunk = id % Kchunks;
  int lane = threadIdx.x;                 // 32 threads
  int hi = lane >> 4, lo = lane & 15;
  int tile = id / Kchunks;
  int n = tile * 16 + lo;
  _Float16* dst = Bop + (size_t)id * 512 + lane * 16;
#pragma unroll
  for (int t = 0; t < 16; ++t) {
    int k = kmap(t, hi) + chunk * 32;
    float v = (k < Krows && n < Ncols) ? W[k * Ncols + n] : 0.0f;
    dst[t] = (_Float16)v;
  }
}

// ---------------------------------------------------------------------------
// Embedding gather: h (f32 master) and h16 (f16 shadow for WMMA A-staging)
// ---------------------------------------------------------------------------
__global__ void k_embed(const int* __restrict__ species,
                        const float* __restrict__ emb,
                        float* __restrict__ h, _Float16* __restrict__ h16) {
  int idx = blockIdx.x * blockDim.x + threadIdx.x;   // over B*N*F
  int f = idx & (FF - 1);
  int node = idx >> 7;
  float v = emb[species[node] * FF + f];
  h[idx] = v;
  h16[idx] = (_Float16)v;
}

// ---------------------------------------------------------------------------
// Filter network: Wf[b,i,j,:] = (ssp(rbf@Wc1+bc1)@Wc2+bc2)*mask, stored f16.
// One block per (b,i); 8 waves, wave w owns j = 16w .. 16w+15.
// ---------------------------------------------------------------------------
__global__ void k_wf(const float* __restrict__ coords,
                     const _Float16* __restrict__ c1op, const float* __restrict__ bc1,
                     const _Float16* __restrict__ c2op, const float* __restrict__ bc2,
                     _Float16* __restrict__ Wf) {
  __shared__ _Float16 s_hid[8][16 * FF];
  __shared__ float    s_mask[8][16];

  int bi = blockIdx.x;                    // b*N + i
  int b = bi >> 7, i = bi & (NN - 1);
  int wave = threadIdx.x >> 5, lane = threadIdx.x & 31;
  int hi = lane >> 4, lo = lane & 15;
  int j = wave * 16 + lo;

  const float* cb = coords + (size_t)b * NN * 3;
  float dx = cb[i * 3 + 0] - cb[j * 3 + 0];
  float dy = cb[i * 3 + 1] - cb[j * 3 + 1];
  float dz = cb[i * 3 + 2] - cb[j * 3 + 2];
  float d2 = dx * dx + dy * dy + dz * dz;
  float d = sqrtf(d2 > 0.0f ? d2 : 1.0f);
  float mk = (d < RC && j != i) ? 1.0f : 0.0f;
  if (hi == 0) s_mask[wave][lo] = mk;

  // RBF A-fragment: row M = lo (pair j), K per kmap. Lanes L and L+16 both
  // recompute d for the same j (cheap) and fill complementary K halves.
  v16h a;
#pragma unroll
  for (int t = 0; t < 16; ++t) {
    int k = kmap(t, hi);
    float mu = 0.5f + (float)k * MU_STEP;
    float dd = d - mu;
    a[t] = (_Float16)__expf(-dd * dd * INV2W2);
  }

  // GEMM1: [16x32] x [32x128], bias + ssp -> LDS (f16)
#pragma unroll
  for (int tile = 0; tile < 8; ++tile) {
    float bias = bc1[tile * 16 + lo];
    v8f c;
#pragma unroll
    for (int r = 0; r < 8; ++r) c[r] = bias;
    v16h bm = *(const v16h*)(c1op + (size_t)tile * 512 + lane * 16);
    c = wmma_f16(a, bm, c);
#pragma unroll
    for (int r = 0; r < 8; ++r) {
      int M = r + hi * 8;
      s_hid[wave][M * FF + tile * 16 + lo] = (_Float16)sspf(c[r]);
    }
  }
  __syncthreads();

  // Restage hidden as A for K=128 (4 chunks)
  v16h a2[4];
#pragma unroll
  for (int ch = 0; ch < 4; ++ch)
#pragma unroll
    for (int t = 0; t < 16; ++t)
      a2[ch][t] = s_hid[wave][lo * FF + ch * 32 + kmap(t, hi)];

  // GEMM2: [16x128] x [128x128], bias, mask, store f16
  _Float16* wfbase = Wf + ((size_t)bi * NN + wave * 16) * FF;
#pragma unroll
  for (int tile = 0; tile < 8; ++tile) {
    float bias = bc2[tile * 16 + lo];
    v8f c;
#pragma unroll
    for (int r = 0; r < 8; ++r) c[r] = bias;
#pragma unroll
    for (int ch = 0; ch < 4; ++ch) {
      v16h bm = *(const v16h*)(c2op + ((size_t)tile * 4 + ch) * 512 + lane * 16);
      c = wmma_f16(a2[ch], bm, c);
    }
#pragma unroll
    for (int r = 0; r < 8; ++r) {
      int M = r + hi * 8;
      float mv = s_mask[wave][M];
      wfbase[(size_t)M * FF + tile * 16 + lo] = (_Float16)(c[r] * mv);
    }
  }
}

// ---------------------------------------------------------------------------
// Message gather: m16[b,i,f] = sum_j Wf[b,i,j,f] * h16[b,j,f]   (f32 accum)
// One block (64 threads, packed feature pair each) per (b,i).
// ---------------------------------------------------------------------------
__global__ void k_gather(const _Float16* __restrict__ Wf,
                         const _Float16* __restrict__ h16,
                         _Float16* __restrict__ m16) {
  int bi = blockIdx.x;
  int b = bi >> 7;
  int f = threadIdx.x * 2;               // 64 threads, features f, f+1
  const _Float16* wrow = Wf + (size_t)bi * NN * FF;
  const _Float16* hb = h16 + (size_t)b * NN * FF;
  float a0 = 0.0f, a1 = 0.0f;
  for (int j = 0; j < NN; ++j) {
    __builtin_prefetch(wrow + (j + 8) * FF + f, 0, 0);   // global_prefetch_b8
    v2h wv = *(const v2h*)(wrow + j * FF + f);
    v2h hv = *(const v2h*)(hb + j * FF + f);
    a0 += (float)wv[0] * (float)hv[0];
    a1 += (float)wv[1] * (float)hv[1];
  }
  v2h o; o[0] = (_Float16)a0; o[1] = (_Float16)a1;
  *(v2h*)(m16 + (size_t)bi * FF + f) = o;
}

// ---------------------------------------------------------------------------
// Update net (residual): h += ssp(m@Wu1+bu1)@Wu2+bu2 ; refresh h16 shadow.
// One block per 16-row M-tile; 8 waves = 8 N-tiles; m tile staged to LDS via
// global_load_async_to_lds_b128 (4KB contiguous, 256 lanes x B128).
// ---------------------------------------------------------------------------
__global__ void k_update(const _Float16* __restrict__ m16,
                         const _Float16* __restrict__ u1op, const float* __restrict__ bu1,
                         const _Float16* __restrict__ u2op, const float* __restrict__ bu2,
                         float* __restrict__ h, _Float16* __restrict__ h16) {
  __shared__ _Float16 s_in[16 * FF];
  __shared__ _Float16 s_hid[16 * FF];
  int row0 = blockIdx.x * 16;
  int wave = threadIdx.x >> 5, lane = threadIdx.x & 31;
  int hi = lane >> 4, lo = lane & 15;

  // async-stage the 16x128 f16 m tile (rows are contiguous in global)
  async_copy_b128(&s_in[threadIdx.x * 8], m16 + (size_t)row0 * FF + threadIdx.x * 8);
  wait_async0();
  __syncthreads();

  v16h a[4];
#pragma unroll
  for (int ch = 0; ch < 4; ++ch)
#pragma unroll
    for (int t = 0; t < 16; ++t)
      a[ch][t] = s_in[lo * FF + ch * 32 + kmap(t, hi)];

  int tile = wave;
  float bias = bu1[tile * 16 + lo];
  v8f c;
#pragma unroll
  for (int r = 0; r < 8; ++r) c[r] = bias;
#pragma unroll
  for (int ch = 0; ch < 4; ++ch) {
    v16h bm = *(const v16h*)(u1op + ((size_t)tile * 4 + ch) * 512 + lane * 16);
    c = wmma_f16(a[ch], bm, c);
  }
#pragma unroll
  for (int r = 0; r < 8; ++r)
    s_hid[(r + hi * 8) * FF + tile * 16 + lo] = (_Float16)sspf(c[r]);
  __syncthreads();

#pragma unroll
  for (int ch = 0; ch < 4; ++ch)
#pragma unroll
    for (int t = 0; t < 16; ++t)
      a[ch][t] = s_hid[lo * FF + ch * 32 + kmap(t, hi)];

  bias = bu2[tile * 16 + lo];
#pragma unroll
  for (int r = 0; r < 8; ++r) c[r] = bias;
#pragma unroll
  for (int ch = 0; ch < 4; ++ch) {
    v16h bm = *(const v16h*)(u2op + ((size_t)tile * 4 + ch) * 512 + lane * 16);
    c = wmma_f16(a[ch], bm, c);
  }
#pragma unroll
  for (int r = 0; r < 8; ++r) {
    int M = r + hi * 8;
    size_t gi = (size_t)(row0 + M) * FF + tile * 16 + lo;
    float nv = h[gi] + c[r];
    h[gi] = nv;
    h16[gi] = (_Float16)nv;
  }
}

// ---------------------------------------------------------------------------
// Node pool: c_iso = ssp(h@Wp1+bp1)@Wp2+bp2   -> out[B*N,8]
// Block = 4 waves per 16-row tile; h16 tile async-staged (2 x B128 per lane).
// ---------------------------------------------------------------------------
__global__ void k_pool(const _Float16* __restrict__ h16,
                       const _Float16* __restrict__ p1op, const float* __restrict__ bp1,
                       const _Float16* __restrict__ p2op, const float* __restrict__ bp2,
                       float* __restrict__ out_iso) {
  __shared__ _Float16 s_in[16 * FF];
  __shared__ _Float16 s_hid[16 * 64];
  int row0 = blockIdx.x * 16;
  int wave = threadIdx.x >> 5, lane = threadIdx.x & 31;
  int hi = lane >> 4, lo = lane & 15;

  const _Float16* src = h16 + (size_t)row0 * FF;
  async_copy_b128(&s_in[threadIdx.x * 8], src + threadIdx.x * 8);
  async_copy_b128(&s_in[(threadIdx.x + 128) * 8], src + (threadIdx.x + 128) * 8);
  wait_async0();
  __syncthreads();

  v16h a[4];
#pragma unroll
  for (int ch = 0; ch < 4; ++ch)
#pragma unroll
    for (int t = 0; t < 16; ++t)
      a[ch][t] = s_in[lo * FF + ch * 32 + kmap(t, hi)];

  int tile = wave;  // 4 tiles cover N=64
  float bias = bp1[tile * 16 + lo];
  v8f c;
#pragma unroll
  for (int r = 0; r < 8; ++r) c[r] = bias;
#pragma unroll
  for (int ch = 0; ch < 4; ++ch) {
    v16h bm = *(const v16h*)(p1op + ((size_t)tile * 4 + ch) * 512 + lane * 16);
    c = wmma_f16(a[ch], bm, c);
  }
#pragma unroll
  for (int r = 0; r < 8; ++r)
    s_hid[(r + hi * 8) * 64 + tile * 16 + lo] = (_Float16)sspf(c[r]);
  __syncthreads();

  if (wave == 0) {                         // wave-uniform: EXEC stays full
    v16h a2[2];
#pragma unroll
    for (int ch = 0; ch < 2; ++ch)
#pragma unroll
      for (int t = 0; t < 16; ++t)
        a2[ch][t] = s_hid[lo * 64 + ch * 32 + kmap(t, hi)];
    float b2 = (lo < 8) ? bp2[lo] : 0.0f;
    v8f c2;
#pragma unroll
    for (int r = 0; r < 8; ++r) c2[r] = b2;
#pragma unroll
    for (int ch = 0; ch < 2; ++ch) {
      v16h bm = *(const v16h*)(p2op + (size_t)ch * 512 + lane * 16);
      c2 = wmma_f16(a2[ch], bm, c2);
    }
    if (lo < 8) {
#pragma unroll
      for (int r = 0; r < 8; ++r)
        out_iso[(size_t)(row0 + r + hi * 8) * 8 + lo] = c2[r];
    }
  }
}

// ---------------------------------------------------------------------------
// Edge pre-projections: pre_i = h@Wei + be1 ; pre_j = h@Wej  (f32 outputs)
// ---------------------------------------------------------------------------
__global__ void k_edgepre(const _Float16* __restrict__ h16,
                          const _Float16* __restrict__ eiop,
                          const _Float16* __restrict__ ejop,
                          const float* __restrict__ be1,
                          float* __restrict__ pre_i, float* __restrict__ pre_j) {
  __shared__ _Float16 s_in[16 * FF];
  int row0 = blockIdx.x * 16;
  int wave = threadIdx.x >> 5, lane = threadIdx.x & 31;
  int hi = lane >> 4, lo = lane & 15;

  async_copy_b128(&s_in[threadIdx.x * 8], h16 + (size_t)row0 * FF + threadIdx.x * 8);
  wait_async0();
  __syncthreads();

  v16h a[4];
#pragma unroll
  for (int ch = 0; ch < 4; ++ch)
#pragma unroll
    for (int t = 0; t < 16; ++t)
      a[ch][t] = s_in[lo * FF + ch * 32 + kmap(t, hi)];

  int tile = wave;
  float bias = be1[tile * 16 + lo];
  v8f ci, cj;
#pragma unroll
  for (int r = 0; r < 8; ++r) { ci[r] = bias; cj[r] = 0.0f; }
#pragma unroll
  for (int ch = 0; ch < 4; ++ch) {
    v16h bmi = *(const v16h*)(eiop + ((size_t)tile * 4 + ch) * 512 + lane * 16);
    ci = wmma_f16(a[ch], bmi, ci);
    v16h bmj = *(const v16h*)(ejop + ((size_t)tile * 4 + ch) * 512 + lane * 16);
    cj = wmma_f16(a[ch], bmj, cj);
  }
#pragma unroll
  for (int r = 0; r < 8; ++r) {
    int M = r + hi * 8;
    size_t gi = (size_t)(row0 + M) * FF + tile * 16 + lo;
    pre_i[gi] = ci[r];
    pre_j[gi] = cj[r];
  }
}

// ---------------------------------------------------------------------------
// Edge pool: out[b,i,:16] = sum_j mask(i,j)*(ssp(pre_i[i]+pre_j[j])@We2) + be2*cnt
// One block per (b,i); 8 waves each own 16 j rows; one WMMA chain vs We2.
// ---------------------------------------------------------------------------
__global__ void k_edgepool(const float* __restrict__ coords,
                           const float* __restrict__ pre_i,
                           const float* __restrict__ pre_j,
                           const _Float16* __restrict__ e2op,
                           const float* __restrict__ be2,
                           float* __restrict__ out_aniso) {
  __shared__ float s_mask[NN];
  __shared__ float s_acc[16];

  int bi = blockIdx.x;
  int b = bi >> 7, i = bi & (NN - 1);
  int wave = threadIdx.x >> 5, lane = threadIdx.x & 31;
  int hi = lane >> 4, lo = lane & 15;
  int j = wave * 16 + lo;

  if (threadIdx.x < 16) s_acc[threadIdx.x] = 0.0f;

  const float* cb = coords + (size_t)b * NN * 3;
  float dx = cb[i * 3 + 0] - cb[j * 3 + 0];
  float dy = cb[i * 3 + 1] - cb[j * 3 + 1];
  float dz = cb[i * 3 + 2] - cb[j * 3 + 2];
  float d2 = dx * dx + dy * dy + dz * dz;
  float d = sqrtf(d2 > 0.0f ? d2 : 1.0f);
  float mk = (d < RC && j != i) ? 1.0f : 0.0f;
  if (hi == 0) s_mask[j] = mk;
  __syncthreads();

  // hid A-fragment, computed on the fly: row M=lo (pair j), feature f per kmap
  const float* pi = pre_i + (size_t)bi * FF;
  const float* pjrow = pre_j + ((size_t)(b * NN) + wave * 16 + lo) * FF;
  v16h a[4];
#pragma unroll
  for (int ch = 0; ch < 4; ++ch)
#pragma unroll
    for (int t = 0; t < 16; ++t) {
      int f = ch * 32 + kmap(t, hi);
      a[ch][t] = (_Float16)sspf(pi[f] + pjrow[f]);
    }

  v8f c;
#pragma unroll
  for (int r = 0; r < 8; ++r) c[r] = 0.0f;
#pragma unroll
  for (int ch = 0; ch < 4; ++ch) {
    v16h bm = *(const v16h*)(e2op + (size_t)ch * 512 + lane * 16);
    c = wmma_f16(a[ch], bm, c);
  }

  // masked row-sum: lane holds column n=lo, rows j = wave*16 + r + hi*8
  float partial = 0.0f;
#pragma unroll
  for (int r = 0; r < 8; ++r)
    partial += c[r] * s_mask[wave * 16 + r + hi * 8];
  partial += __shfl_down(partial, 16, 32);       // fold high half-wave rows
  if (hi == 0) atomicAdd(&s_acc[lo], partial);   // ds_add_f32 across 8 waves
  __syncthreads();

  if (threadIdx.x < 16) {
    float cnt = 0.0f;
    for (int jj = 0; jj < NN; ++jj) cnt += s_mask[jj];
    out_aniso[(size_t)bi * 16 + threadIdx.x] =
        s_acc[threadIdx.x] + be2[threadIdx.x] * cnt;
  }
}

// ---------------------------------------------------------------------------
// Host launcher
// ---------------------------------------------------------------------------
extern "C" void kernel_launch(void* const* d_in, const int* in_sizes, int n_in,
                              void* d_out, int out_size, void* d_ws, size_t ws_size,
                              hipStream_t stream) {
  (void)in_sizes; (void)n_in; (void)out_size; (void)ws_size;

  const float* coords  = (const float*)d_in[0];
  const int*   species = (const int*)d_in[1];
  const float* emb     = (const float*)d_in[2];
  const float* Wc1 = (const float*)d_in[3];  const float* bc1 = (const float*)d_in[4];
  const float* Wc2 = (const float*)d_in[5];  const float* bc2 = (const float*)d_in[6];
  const float* Wu1 = (const float*)d_in[7];  const float* bu1 = (const float*)d_in[8];
  const float* Wu2 = (const float*)d_in[9];  const float* bu2 = (const float*)d_in[10];
  const float* Wp1 = (const float*)d_in[11]; const float* bp1 = (const float*)d_in[12];
  const float* Wp2 = (const float*)d_in[13]; const float* bp2 = (const float*)d_in[14];
  const float* Wei = (const float*)d_in[15]; const float* Wej = (const float*)d_in[16];
  const float* be1 = (const float*)d_in[17];
  const float* We2 = (const float*)d_in[18]; const float* be2 = (const float*)d_in[19];

  float* out_iso   = (float*)d_out;                       // [B*N, 8]
  float* out_aniso = (float*)d_out + (size_t)BB * NN * 8; // [B*N, 16]

  // ---- workspace carving (assumes ws_size >= ~70 MB) ----
  char* ws = (char*)d_ws;
  size_t off = 0;
  auto carve = [&](size_t bytes) -> char* {
    char* p = ws + off;
    off = (off + bytes + 255) & ~(size_t)255;
    return p;
  };
  float*    h     = (float*)carve((size_t)BB * NN * FF * 4);
  _Float16* h16   = (_Float16*)carve((size_t)BB * NN * FF * 2);
  _Float16* m16   = (_Float16*)carve((size_t)BB * NN * FF * 2);
  float*    pre_i = (float*)carve((size_t)BB * NN * FF * 4);
  float*    pre_j = (float*)carve((size_t)BB * NN * FF * 4);
  _Float16* c1op = (_Float16*)carve((size_t)8  * 512 * 2);  // 8 tiles x 1 chunk
  _Float16* c2op = (_Float16*)carve((size_t)32 * 512 * 2);  // 8 x 4
  _Float16* u1op = (_Float16*)carve((size_t)32 * 512 * 2);
  _Float16* u2op = (_Float16*)carve((size_t)32 * 512 * 2);
  _Float16* p1op = (_Float16*)carve((size_t)16 * 512 * 2);  // 4 x 4
  _Float16* p2op = (_Float16*)carve((size_t)2  * 512 * 2);  // 1 x 2 (N padded)
  _Float16* eiop = (_Float16*)carve((size_t)32 * 512 * 2);
  _Float16* ejop = (_Float16*)carve((size_t)32 * 512 * 2);
  _Float16* e2op = (_Float16*)carve((size_t)4  * 512 * 2);  // 1 x 4
  _Float16* Wf   = (_Float16*)carve((size_t)BB * NN * NN * FF * 2); // 64MB, L2-resident

  // ---- weight repack to WMMA B-operand images ----
  k_prep_bop<<<8,  32, 0, stream>>>(Wc1, c1op, KK,  FF, 1);
  k_prep_bop<<<32, 32, 0, stream>>>(Wc2, c2op, FF,  FF, 4);
  k_prep_bop<<<32, 32, 0, stream>>>(Wu1, u1op, FF,  FF, 4);
  k_prep_bop<<<32, 32, 0, stream>>>(Wu2, u2op, FF,  FF, 4);
  k_prep_bop<<<16, 32, 0, stream>>>(Wp1, p1op, FF,  64, 4);
  k_prep_bop<<<2,  32, 0, stream>>>(Wp2, p2op, 64,  8,  2);
  k_prep_bop<<<32, 32, 0, stream>>>(Wei, eiop, FF,  FF, 4);
  k_prep_bop<<<32, 32, 0, stream>>>(Wej, ejop, FF,  FF, 4);
  k_prep_bop<<<4,  32, 0, stream>>>(We2, e2op, FF,  16, 4);

  // ---- embedding ----
  k_embed<<<(BB * NN * FF) / 256, 256, 0, stream>>>(species, emb, h, h16);

  // ---- continuous-filter weights (the 10.7 GFLOP WMMA hot spot) ----
  k_wf<<<BB * NN, 256, 0, stream>>>(coords, c1op, bc1, c2op, bc2, Wf);

  // ---- message passing x3 ----
  for (int it = 0; it < N_CONV; ++it) {
    k_gather<<<BB * NN, 64, 0, stream>>>(Wf, h16, m16);
    k_update<<<(BB * NN) / 16, 256, 0, stream>>>(m16, u1op, bu1, u2op, bu2, h, h16);
  }

  // ---- node pool -> c_iso ----
  k_pool<<<(BB * NN) / 16, 128, 0, stream>>>(h16, p1op, bp1, p2op, bp2, out_iso);

  // ---- edge pool -> c_aniso ----
  k_edgepre<<<(BB * NN) / 16, 256, 0, stream>>>(h16, eiop, ejop, be1, pre_i, pre_j);
  k_edgepool<<<BB * NN, 256, 0, stream>>>(coords, pre_i, pre_j, e2op, be2, out_aniso);
}
